// MambaBlock_37082747633911
// MI455X (gfx1250) — matrix operand order repaired and checked by reading, not compiled, not run on hardware
//
#include <hip/hip_runtime.h>
#include <hip/hip_bf16.h>

typedef __attribute__((ext_vector_type(16))) _Float16 v16h;
typedef __attribute__((ext_vector_type(8)))  _Float16 v8h;
typedef __attribute__((ext_vector_type(8)))  float    v8f;
typedef __attribute__((ext_vector_type(4)))  int      v4i;

#define DIMX    1024
#define DSTATE  16
#define DCONV   4
#define DINNER  2048
#define NB      2
#define NL      1024
#define NTOK    (NB * NL)      // 2048 tokens

#define LDA 72                 // LDS row stride in halves (32 data + 40 pad) -> 144B, 36 banks

#if __has_builtin(__builtin_amdgcn_global_load_async_to_lds_b128)
#define HAVE_ASYNC_COPY 1
#else
#define HAVE_ASYNC_COPY 0
#endif
#if __has_builtin(__builtin_amdgcn_s_wait_asynccnt)
#define HAVE_WAIT_ASYNC 1
#else
#define HAVE_WAIT_ASYNC 0
#endif

typedef __attribute__((address_space(1))) v4i* g_v4i_p;   // global int4*
typedef __attribute__((address_space(3))) v4i* l_v4i_p;   // LDS int4*

// one 16-byte global -> LDS transfer
__device__ __forceinline__ void copy16_g2l(const _Float16* g, _Float16* l) {
#if HAVE_ASYNC_COPY
    __builtin_amdgcn_global_load_async_to_lds_b128((g_v4i_p)g, (l_v4i_p)l, 0, 0);
#else
    *(float4*)l = *(const float4*)g;
#endif
}

template<int N> __device__ __forceinline__ void wait_async() {
#if HAVE_ASYNC_COPY
#if HAVE_WAIT_ASYNC
    __builtin_amdgcn_s_wait_asynccnt(N);
#else
    asm volatile("s_wait_asynccnt %0" :: "n"(N));
#endif
#endif
}

// ---------------------------------------------------------------- cast f32->f16
__global__ __launch_bounds__(256) void cast_f32_f16(const float* __restrict__ s,
                                                    _Float16* __restrict__ d, int n) {
    int i = (blockIdx.x * blockDim.x + threadIdx.x) * 4;
    if (i + 3 < n) {
        float4 v = *(const float4*)(s + i);
        d[i + 0] = (_Float16)v.x;
        d[i + 1] = (_Float16)v.y;
        d[i + 2] = (_Float16)v.z;
        d[i + 3] = (_Float16)v.w;
    }
}

// ---------------------------------------------------------------- WMMA GEMM (NT)
// C[M,N] = A[M,K] * B[N,K]^T, f16 in, f32 accumulate.
// Block: 256 thr (8 waves), tile 128M x 64N; per wave 32x32 (2x2 WMMA of 16x16x32).
// A/B tiles double-buffered in LDS, filled via async global->LDS b128 transfers.
// EPI: 0 = plain store, 1 = softplus(acc + bias[n]), 2 = acc + resid[m,n]
template<int EPI>
__global__ __launch_bounds__(256)
void gemm_nt_wmma(const _Float16* __restrict__ A, const _Float16* __restrict__ Bm,
                  float* __restrict__ C, int M, int N, int K,
                  const float* __restrict__ bias, const float* __restrict__ resid) {
    __shared__ _Float16 As[2][128 * LDA];
    __shared__ _Float16 Bs[2][64 * LDA];

    const int t    = threadIdx.x;
    const int lane = t & 31;
    const int wave = t >> 5;
    const int mw = wave & 3;                 // 4 M-subtiles of 32
    const int nw = wave >> 2;                // 2 N-subtiles of 32
    const int mBase0 = blockIdx.x * 128;
    const int nBase0 = blockIdx.y * 64;

    // fragment addressing within LDS tile (ISA VGPR layouts)
    const int am    = lane & 15;
    const int akoff = (lane >> 4) << 3;      // 0 or 8
    const int bn    = lane & 15;
    const int bkoff = (lane >> 4) << 3;      // 0 or 8
    const int cm    = (lane >> 4) << 3;
    const int cn    = lane & 15;

    // stage loader: A tile 128x32 halves = 512 x 16B, B tile 64x32 = 256 x 16B
    auto load_stage = [&](int k0, int buf) {
#pragma unroll
        for (int h = 0; h < 2; ++h) {
            int i   = t + h * 256;
            int row = i >> 2;
            int ch  = i & 3;
            copy16_g2l(A + (size_t)(mBase0 + row) * K + k0 + ch * 8,
                       &As[buf][row * LDA + ch * 8]);
        }
        {
            int row = t >> 2;
            int ch  = t & 3;
            copy16_g2l(Bm + (size_t)(nBase0 + row) * K + k0 + ch * 8,
                       &Bs[buf][row * LDA + ch * 8]);
        }
    };

    v8f acc[2][2] = {};
    const int nst = K >> 5;                  // K / 32 stages
    load_stage(0, 0);

    for (int s = 0; s < nst; ++s) {
        const int buf = s & 1;
        if (s + 1 < nst) {
            load_stage((s + 1) << 5, buf ^ 1);
            wait_async<3>();                 // current stage's 3 transfers landed
        } else {
            wait_async<0>();
        }
        __syncthreads();                     // stage visible to all waves

        v16h a[2], b[2];
#pragma unroll
        for (int i = 0; i < 2; ++i) {
            const _Float16* pa = &As[buf][(mw * 32 + i * 16 + am) * LDA + akoff];
            v8h alo = *(const v8h*)pa;       // ds_load_b128
            v8h ahi = *(const v8h*)(pa + 16);
            a[i] = __builtin_shufflevector(alo, ahi, 0, 1, 2, 3, 4, 5, 6, 7,
                                           8, 9, 10, 11, 12, 13, 14, 15);
            const _Float16* pb = &Bs[buf][(nw * 32 + i * 16 + bn) * LDA + bkoff];
            v8h blo = *(const v8h*)pb;
            v8h bhi = *(const v8h*)(pb + 16);
            b[i] = __builtin_shufflevector(blo, bhi, 0, 1, 2, 3, 4, 5, 6, 7,
                                           8, 9, 10, 11, 12, 13, 14, 15);
        }
#pragma unroll
        for (int i = 0; i < 2; ++i)
#pragma unroll
            for (int j = 0; j < 2; ++j)
                acc[i][j] = __builtin_amdgcn_wmma_f32_16x16x32_f16(
                    false, a[i], false, b[j], (short)0, acc[i][j], false, false);

        __syncthreads();                     // all reads done before buf is refilled
    }

    // C/D layout: element (r,lane) -> m = r + (lane>>4)*8, n = lane&15
#pragma unroll
    for (int i = 0; i < 2; ++i)
#pragma unroll
        for (int j = 0; j < 2; ++j)
#pragma unroll
            for (int r = 0; r < 8; ++r) {
                int row = mBase0 + mw * 32 + i * 16 + cm + r;
                int col = nBase0 + nw * 32 + j * 16 + cn;
                float v = acc[i][j][r];
                if (EPI == 1) {              // softplus(acc + bias)
                    v += bias[col];
                    v = (v > 20.f) ? v : logf(1.f + __expf(v));
                }
                if (EPI == 2)                // residual add
                    v += resid[(size_t)row * N + col];
                C[(size_t)row * N + col] = v;
            }
}

// ------------------------------------------- causal depthwise conv1d + bias + SiLU
__global__ __launch_bounds__(256) void conv_silu(const float* __restrict__ xz,
                                                 const float* __restrict__ Wc,
                                                 const float* __restrict__ bc,
                                                 float* __restrict__ xci,
                                                 _Float16* __restrict__ xcih) {
    int idx = blockIdx.x * blockDim.x + threadIdx.x;   // over NTOK*DINNER
    if (idx >= NTOK * DINNER) return;
    int c = idx & (DINNER - 1);
    int n = idx >> 11;          // token index (DINNER = 2^11)
    int l = n & (NL - 1);
    float acc = bc[c];
#pragma unroll
    for (int j = 0; j < DCONV; ++j) {
        int li = l - (DCONV - 1) + j;
        if (li >= 0)
            acc += Wc[c * DCONV + j] *
                   xz[(size_t)(n - (DCONV - 1) + j) * (2 * DINNER) + c];
    }
    float s = acc / (1.f + __expf(-acc));   // silu
    xci[idx]  = s;
    xcih[idx] = (_Float16)s;
}

// ------------------------------------------- B_t / C_t projections (N = 16 each)
__global__ __launch_bounds__(256) void bc_proj(const float* __restrict__ xci,
                                               const float* __restrict__ WB,
                                               const float* __restrict__ WC,
                                               float* __restrict__ Bt,
                                               float* __restrict__ Ct) {
    __shared__ float row[DINNER];
    int n = blockIdx.x;
    for (int i = threadIdx.x; i < DINNER; i += 256)
        row[i] = xci[(size_t)n * DINNER + i];
    __syncthreads();
    int out  = threadIdx.x >> 3;     // 32 outputs (16 B + 16 C)
    int part = threadIdx.x & 7;      // 8 lanes per output
    const float* W = (out < 16) ? WB : WC;
    int s = out & 15;
    float acc = 0.f;
    for (int k = part; k < DINNER; k += 8)
        acc += row[k] * W[s * DINNER + k];
    acc += __shfl_xor(acc, 1);
    acc += __shfl_xor(acc, 2);
    acc += __shfl_xor(acc, 4);
    if (part == 0) {
        if (out < 16) Bt[n * DSTATE + s] = acc;
        else          Ct[n * DSTATE + s] = acc;
    }
}

// ------------------------------------------- selective scan, fused with SiLU(z) gate
// lane s = state index (16 lanes / channel), wave32 handles 2 channels
__global__ __launch_bounds__(256) void scan_fused(const float* __restrict__ delta,
                                                  const float* __restrict__ xci,
                                                  const float* __restrict__ Bt,
                                                  const float* __restrict__ Ct,
                                                  const float* __restrict__ A_log,
                                                  const float* __restrict__ Dp,
                                                  const float* __restrict__ xz,
                                                  _Float16* __restrict__ yh) {
    int lane = threadIdx.x & 31;
    int wave = threadIdx.x >> 5;
    int s  = lane & 15;
    int cp = blockIdx.x * 8 + wave;          // channel pair 0..2047
    int cg = cp * 2 + (lane >> 4);           // global (b,c) 0..4095
    int b  = cg >> 11;
    int c  = cg & (DINNER - 1);
    float Aa = -__expf(A_log[s]);
    float Dc = Dp[c];
    float h  = 0.f;
    for (int l = 0; l < NL; ++l) {
        size_t nrow = (size_t)(b * NL + l);
        float dlt = delta[nrow * DINNER + c];
        float xv  = xci[nrow * DINNER + c];
        float Bv  = Bt[nrow * DSTATE + s];
        float Cv  = Ct[nrow * DSTATE + s];
        h = __expf(dlt * Aa) * h + dlt * Bv * xv;
        float p = h * Cv;
        p += __shfl_xor(p, 1);
        p += __shfl_xor(p, 2);
        p += __shfl_xor(p, 4);
        p += __shfl_xor(p, 8);
        if (s == 0) {
            float zv = xz[nrow * (2 * DINNER) + DINNER + c];
            float y  = (p + Dc * xv) * (zv / (1.f + __expf(-zv)));
            yh[nrow * DINNER + c] = (_Float16)y;
        }
    }
}

// ---------------------------------------------------------------- launch
extern "C" void kernel_launch(void* const* d_in, const int* in_sizes, int n_in,
                              void* d_out, int out_size, void* d_ws, size_t ws_size,
                              hipStream_t stream) {
    const float* x      = (const float*)d_in[0];   // [2,1024,1024]
    const float* W_in   = (const float*)d_in[1];   // [4096,1024]
    const float* W_conv = (const float*)d_in[2];   // [2048,1,4]
    const float* b_conv = (const float*)d_in[3];   // [2048]
    const float* W_dt   = (const float*)d_in[4];   // [2048,2048]
    const float* b_dt   = (const float*)d_in[5];   // [2048]
    const float* W_B    = (const float*)d_in[6];   // [16,2048]
    const float* W_C    = (const float*)d_in[7];   // [16,2048]
    const float* A_log  = (const float*)d_in[8];   // [16]
    const float* Dp     = (const float*)d_in[9];   // [2048]
    const float* W_out  = (const float*)d_in[10];  // [1024,2048]
    float* out = (float*)d_out;

    char* ws = (char*)d_ws;
    _Float16* xh    = (_Float16*)(ws);                          //  4 MiB
    _Float16* Winh  = (_Float16*)(ws + ((size_t)4 << 20));      //  8 MiB
    float*    xz    = (float*)   (ws + ((size_t)12 << 20));     // 32 MiB
    float*    xci   = (float*)   (ws + ((size_t)44 << 20));     // 16 MiB
    _Float16* xcih  = (_Float16*)(ws + ((size_t)60 << 20));     //  8 MiB
    _Float16* Wdth  = (_Float16*)(ws + ((size_t)68 << 20));     //  8 MiB
    float*    delta = (float*)   (ws + ((size_t)76 << 20));     // 16 MiB
    float*    Bt    = (float*)   (ws + ((size_t)92 << 20));     // 128 KiB
    float*    Ct    = (float*)   (ws + ((size_t)93 << 20));     // 128 KiB
    _Float16* yh    = (_Float16*)(ws + ((size_t)94 << 20));     //  8 MiB
    _Float16* Wouth = (_Float16*)(ws + ((size_t)102 << 20));    //  4 MiB

    // 1) f16 conversions
    cast_f32_f16<<<(NTOK * DIMX / 4 + 255) / 256, 256, 0, stream>>>(x, xh, NTOK * DIMX);
    cast_f32_f16<<<(2 * DINNER * DIMX / 4 + 255) / 256, 256, 0, stream>>>(W_in, Winh, 2 * DINNER * DIMX);
    cast_f32_f16<<<(DINNER * DINNER / 4 + 255) / 256, 256, 0, stream>>>(W_dt, Wdth, DINNER * DINNER);
    cast_f32_f16<<<(DIMX * DINNER / 4 + 255) / 256, 256, 0, stream>>>(W_out, Wouth, DIMX * DINNER);

    // 2) in_proj: xz[2048,4096] = x * W_in^T
    gemm_nt_wmma<0><<<dim3(NTOK / 128, (2 * DINNER) / 64), 256, 0, stream>>>(
        xh, Winh, xz, NTOK, 2 * DINNER, DIMX, nullptr, nullptr);

    // 3) causal depthwise conv + SiLU
    conv_silu<<<(NTOK * DINNER + 255) / 256, 256, 0, stream>>>(xz, W_conv, b_conv, xci, xcih);

    // 4) dt_proj + softplus: delta[2048,2048]
    gemm_nt_wmma<1><<<dim3(NTOK / 128, DINNER / 64), 256, 0, stream>>>(
        xcih, Wdth, delta, NTOK, DINNER, DINNER, b_dt, nullptr);

    // 5) B_t / C_t projections
    bc_proj<<<NTOK, 256, 0, stream>>>(xci, W_B, W_C, Bt, Ct);

    // 6) selective scan fused with SiLU(z) gate -> y (f16)
    scan_fused<<<(NB * DINNER / 2) / 8, 256, 0, stream>>>(delta, xci, Bt, Ct, A_log, Dp, xz, yh);

    // 7) out_proj + residual: out = x + y * W_out^T
    gemm_nt_wmma<2><<<dim3(NTOK / 128, DIMX / 64), 256, 0, stream>>>(
        yh, Wouth, out, NTOK, DIMX, DINNER, nullptr, x);
}
